// HeteAttention_35450660061722
// MI455X (gfx1250) — matrix operand order repaired and checked by reading, not compiled, not run on hardware
//
#include <hip/hip_runtime.h>
#include <hip/hip_bf16.h>
#include <math.h>

// ---------------------------------------------------------------------------
// HeteAttention facet-attention kernel for gfx1250 (MI455X)
//   B=2048, NFACET=8, D=64, NB=64, N_ITER=3
//   One workgroup (256 threads = 8 wave32) per batch element.
//   Wave w owns facet f=w. z[b] lives in LDS (f16, dual layout) for all iters.
//   GEMVs run on v_wmma_f32_16x16x32_f16 with the vector operand broadcast
//   across all 16 B-columns (every output column == GEMV result).
//   Write-back uses branch-free v_cndmask selection (no saveexec chains).
// ---------------------------------------------------------------------------

#define NFACET 8
#define DIM    64
#define NBR    64
#define FD     (NFACET * DIM)          // 512
#define ZEL    (NFACET * NBR * DIM)    // 32768 elements of z per batch item
#define NITER  3

typedef __attribute__((ext_vector_type(16))) _Float16 v16h;
typedef __attribute__((ext_vector_type(8)))  _Float16 v8h_t;
typedef __attribute__((ext_vector_type(4)))  _Float16 v4h_t;
typedef __attribute__((ext_vector_type(8)))  float    v8f;

// A-operand tile load, 16-bit A 16x32 layout (ISA 7.12.2):
// lane group kb = 0 (lanes 0-15) or 8 (lanes 16-31);
// elements 0..7  = row[kb .. kb+7], elements 8..15 = row[kb+16 .. kb+23]
static __device__ __forceinline__ v16h ld_a16(const _Float16* p) {
    v8h_t lo = *(const v8h_t*)(p);
    v8h_t hi = *(const v8h_t*)(p + 16);
    return __builtin_shufflevector(lo, hi, 0, 1, 2, 3, 4, 5, 6, 7,
                                           8, 9, 10, 11, 12, 13, 14, 15);
}

// B-operand (broadcast vector): elements 0..7 = p[0..7], 8..15 = p[16..23]
static __device__ __forceinline__ v16h ld_b16(const _Float16* p) {
    v8h_t lo = *(const v8h_t*)(p);
    v8h_t hi = *(const v8h_t*)(p + 16);
    return __builtin_shufflevector(lo, hi, 0, 1, 2, 3, 4, 5, 6, 7,
                                           8, 9, 10, 11, 12, 13, 14, 15);
}

// Branch-free extraction of element `c` (0..7): 7 v_cndmask, no exec masking.
static __device__ __forceinline__ float sel8(v8f a, int c) {
    float v = a[0];
    #pragma unroll
    for (int i = 1; i < 8; ++i) v = (c == i) ? a[i] : v;
    return v;
}

__global__ __launch_bounds__(256)
void hete_attention_gfx1250(const float* __restrict__ feat,
                            const float* __restrict__ mp,
                            float* __restrict__ out)
{
    extern __shared__ char smem[];
    _Float16* z16  = (_Float16*)smem;            // [F][NB][D]  65536 B
    _Float16* zt16 = z16 + ZEL;                  // [F][D][NB]  65536 B
    float*    x32  = (float*)(zt16 + ZEL);       // [F][D]       2048 B
    float*    p32  = x32 + FD;                   // [NB][F]      2048 B
    float*    sums = p32 + FD;                   // [NB][F]      2048 B
    _Float16* u16  = (_Float16*)(sums + FD);     // [F][D]       1024 B
    _Float16* p16  = u16 + FD;                   // [F][NB]      1024 B

    const int b    = blockIdx.x;
    const int t    = threadIdx.x;
    const int lane = t & 31;
    const int wv   = t >> 5;          // facet owned by this wave
    const int hi   = lane >> 4;       // half-wave id
    const int c    = lane & 15;       // column id (== selected row within half)
    const int kbA  = hi ? 8 : 0;      // A-operand K base
    const int kbB  = hi ? 8 : 0;      // B-operand K base
    const bool wr  = (c < 8);         // lanes that perform write-back

    // ---- zero sumsq accumulators ------------------------------------------
    if (t < 128) {
        #pragma unroll
        for (int j = 0; j < 4; ++j) sums[t * 4 + j] = 0.0f;
    }

    // ---- x init: normalize features, seed u -------------------------------
    {
        const float* xr = feat + (size_t)b * FD + wv * DIM;
        float x0 = xr[lane];
        float x1 = xr[lane + 32];
        float s  = x0 * x0 + x1 * x1;
        #pragma unroll
        for (int m = 16; m >= 1; m >>= 1) s += __shfl_xor(s, m, 32);
        float inv = 1.0f / fmaxf(sqrtf(s), 1e-12f);
        x0 *= inv; x1 *= inv;
        x32[wv * DIM + lane]      = x0;
        x32[wv * DIM + lane + 32] = x1;
        u16[wv * DIM + lane]      = (_Float16)x0;
        u16[wv * DIM + lane + 32] = (_Float16)x1;
    }
    __syncthreads();

    // ---- pass 1: coalesced HBM -> LDS (f16), accumulate row sumsq ---------
    {
        const float4* src = (const float4*)(mp + (size_t)b * ZEL);
        #pragma unroll 4
        for (int i = 0; i < ZEL / 4 / 256; ++i) {        // 32 iterations
            int    idx  = i * 256 + t;
            float4 v    = src[idx];
            int    flat = idx * 4;
            int    n    = flat >> 9;          // / 512
            int    r    = flat & 511;
            int    f    = r >> 6;             // / 64
            int    d    = r & 63;
            float  s    = v.x * v.x + v.y * v.y + v.z * v.z + v.w * v.w;
            atomicAdd(&sums[n * NFACET + f], s);
            v4h_t h;
            h[0] = (_Float16)v.x; h[1] = (_Float16)v.y;
            h[2] = (_Float16)v.z; h[3] = (_Float16)v.w;
            *(v4h_t*)(z16 + f * NBR * DIM + n * DIM + d) = h;
        }
    }
    __syncthreads();

    // ---- pass 2: normalize rows in place, build transposed copy -----------
    #pragma unroll
    for (int j = 0; j < 2; ++j) {
        int   r   = t + j * 256;              // (f*64 + n)
        int   f   = r >> 6;
        int   n   = r & 63;
        float s   = sums[n * NFACET + f];
        float inv = 1.0f / fmaxf(sqrtf(s), 1e-12f);
        _Float16* row = z16  + f * NBR * DIM + n * DIM;
        _Float16* col = zt16 + f * NBR * DIM + n;
        for (int d = 0; d < DIM; ++d) {
            _Float16 h = (_Float16)((float)row[d] * inv);
            row[d]       = h;
            col[d * DIM] = h;                 // zt16[f][d][n]
        }
    }
    __syncthreads();

    // ---- iteration loop ----------------------------------------------------
    const _Float16* zf  = z16  + wv * NBR * DIM;
    const _Float16* ztf = zt16 + wv * NBR * DIM;

    for (int it = 0; it < NITER; ++it) {
        // === GEMV1: p = Z_f @ u  (8 wmma) ==================================
        {
            const _Float16* uf = u16 + wv * DIM;
            v16h b0 = ld_b16(uf + kbB);           // K = 0..31 slice
            v16h b1 = ld_b16(uf + 32 + kbB);      // K = 32..63 slice
            #pragma unroll
            for (int tt = 0; tt < 4; ++tt) {
                const _Float16* zr = zf + (tt * 16 + c) * DIM;
                v16h a0 = ld_a16(zr + kbA);
                v16h a1 = ld_a16(zr + 32 + kbA);
                v8f  acc = {};
                acc = __builtin_amdgcn_wmma_f32_16x16x32_f16(
                          false, a0, false, b0, (short)0, acc, false, false);
                acc = __builtin_amdgcn_wmma_f32_16x16x32_f16(
                          false, a1, false, b1, (short)0, acc, false, false);
                // lane with column id c writes row (tt*16 + hi*8 + c)
                float pv = sel8(acc, c);
                if (wr) p32[(tt * 16 + hi * 8 + c) * NFACET + wv] = pv;
            }
        }
        __syncthreads();

        // === softmax over facets at fixed (b,n) ============================
        #pragma unroll
        for (int j = 0; j < 2; ++j) {
            int n = lane + j * 32;
            const float* pr = p32 + n * NFACET;
            float mx = pr[0];
            #pragma unroll
            for (int f = 1; f < NFACET; ++f) mx = fmaxf(mx, pr[f]);
            float e[NFACET], sum = 0.0f;
            #pragma unroll
            for (int f = 0; f < NFACET; ++f) { e[f] = __expf(pr[f] - mx); sum += e[f]; }
            p16[wv * NBR + n] = (_Float16)(e[wv] / sum);
        }
        __syncthreads();

        // === GEMV2: u' = Z_f^T @ p + x  (8 wmma) ===========================
        {
            const _Float16* pf = p16 + wv * NBR;
            v16h b0 = ld_b16(pf + kbB);
            v16h b1 = ld_b16(pf + 32 + kbB);
            v8f  accs[4];
            #pragma unroll
            for (int tt = 0; tt < 4; ++tt) {
                const _Float16* zr = ztf + (tt * 16 + c) * DIM;
                v16h a0 = ld_a16(zr + kbA);
                v16h a1 = ld_a16(zr + 32 + kbA);
                v8f  acc = {};
                acc = __builtin_amdgcn_wmma_f32_16x16x32_f16(
                          false, a0, false, b0, (short)0, acc, false, false);
                acc = __builtin_amdgcn_wmma_f32_16x16x32_f16(
                          false, a1, false, b1, (short)0, acc, false, false);
                accs[tt] = acc;
            }
            // add residual x, accumulate partial sumsq (cols are replicated)
            float part = 0.0f;
            #pragma unroll
            for (int tt = 0; tt < 4; ++tt) {
                v8f xv = *(const v8f*)(x32 + wv * DIM + tt * 16 + hi * 8);
                #pragma unroll
                for (int v = 0; v < 8; ++v) {
                    float val = accs[tt][v] + xv[v];
                    accs[tt][v] = val;
                    part += val * val;
                }
            }
            float total = part + __shfl_xor(part, 16, 32);
            if (it < NITER - 1) {
                float inv = 1.0f / fmaxf(sqrtf(total), 1e-12f);
                #pragma unroll
                for (int tt = 0; tt < 4; ++tt) {
                    float val = sel8(accs[tt], c) * inv;
                    if (wr) u16[wv * DIM + tt * 16 + hi * 8 + c] = (_Float16)val;
                }
            } else {
                float* orow = out + (size_t)b * FD + wv * DIM;
                #pragma unroll
                for (int tt = 0; tt < 4; ++tt) {
                    float val = sel8(accs[tt], c);
                    if (wr) orow[tt * 16 + hi * 8 + c] = val;
                }
            }
        }
        __syncthreads();
    }
}

extern "C" void kernel_launch(void* const* d_in, const int* in_sizes, int n_in,
                              void* d_out, int out_size, void* d_ws, size_t ws_size,
                              hipStream_t stream) {
    const float* feat = (const float*)d_in[0];   // [B, 512]
    const float* mp   = (const float*)d_in[1];   // [B, 64, 512]
    float*       out  = (float*)d_out;           // [B, 512]

    const int B = in_sizes[0] / FD;              // 2048
    const size_t smem = (size_t)(2 * ZEL) * sizeof(_Float16)   // z16 + zt16
                      + (size_t)(3 * FD) * sizeof(float)       // x32, p32, sums
                      + (size_t)(2 * FD) * sizeof(_Float16);   // u16, p16

    // Allow >64KB dynamic LDS (host-side attribute; not a stream op, so
    // graph-capture safe; deterministic every call).
    (void)hipFuncSetAttribute((const void*)hete_attention_gfx1250,
                              hipFuncAttributeMaxDynamicSharedMemorySize,
                              (int)smem);

    hete_attention_gfx1250<<<dim3(B), dim3(256), smem, stream>>>(feat, mp, out);
}
